// MaskedSelfAttention_84069689852622
// MI455X (gfx1250) — matrix operand order repaired and checked by reading, not compiled, not run on hardware
//
#include <hip/hip_runtime.h>
#include <hip/hip_bf16.h>

// ---- CDNA5 WMMA / TDM vector types ----
typedef __attribute__((ext_vector_type(16))) __bf16       v16bf;
typedef __attribute__((ext_vector_type(8)))  __bf16       v8bf;
typedef __attribute__((ext_vector_type(8)))  float        v8f;
typedef __attribute__((ext_vector_type(4)))  unsigned int v4u;
typedef __attribute__((ext_vector_type(8)))  int          v8i;
typedef __attribute__((ext_vector_type(4)))  int          v4i;

#define S_LEN  2048
#define DMODEL 1024
#define DH     64
#define NBATCH 4

#if defined(__gfx1250__) && __has_builtin(__builtin_amdgcn_tensor_load_to_lds)
#define USE_TDM 1
#else
#define USE_TDM 0
#endif

#if USE_TDM
// Build a 2D-tile Tensor DMA descriptor (D#) and issue TENSOR_LOAD_TO_LDS.
// Layout per CDNA5 ISA. data_size = 2 bytes (bf16).
// This toolchain exposes the 6-arg builtin form:
//   (uint32x4 g0, int32x8 g1, int32x4 g2, int32x4 g3, int32x8 extra, i32 cpol)
// Groups 2/3 (+extra) are zero for 2D tensors. All args wave-uniform -> SGPRs.
__device__ __forceinline__ void tdm_load_tile_2d(
    unsigned lds_byte_addr, const void* global_tile_addr,
    unsigned tensor_d0, unsigned tensor_d1,
    unsigned tile_d0, unsigned tile_d1, unsigned stride0_elems)
{
    unsigned long long ga = (unsigned long long)global_tile_addr;
    v4u g0 = { 1u,                                  // [1:0] count=1, user mode
               lds_byte_addr,                       // [63:32] lds_addr
               (unsigned)ga,                        // global_addr[31:0]
               ((unsigned)(ga >> 32) & 0x1ffffffu)  // global_addr[56:32]
                   | (2u << 30) };                  // [127:126] type=2 (image)
    v8i g1 = { (int)(1u << 16),                     // data_size=1 (2 bytes)
               (int)((tensor_d0 & 0xffffu) << 16),  // tensor_dim0[15:0]
               (int)(((tensor_d0 >> 16) & 0xffffu) | ((tensor_d1 & 0xffffu) << 16)),
               (int)(((tensor_d1 >> 16) & 0xffffu) | ((tile_d0 & 0xffffu) << 16)),
               (int)(tile_d1 & 0xffffu),            // tile_dim1 (tile_dim2=0)
               (int)stride0_elems,                  // tensor_dim0_stride[31:0]
               0, 0 };
    v4i zero4 = {0, 0, 0, 0};
    v8i zero8 = {0, 0, 0, 0, 0, 0, 0, 0};
    __builtin_amdgcn_tensor_load_to_lds(g0, g1, zero4, zero4, zero8, 0);
}
#endif

// =====================================================================
// Kernel 0: one-off weight transpose+convert: Wq|Wk|Wv [1024,64] fp32
//           -> WT [3][64][1024] bf16 (makes all B-fragments contiguous).
// =====================================================================
__global__ void wtrans_kernel(const float* __restrict__ Wq,
                              const float* __restrict__ Wk,
                              const float* __restrict__ Wv,
                              __bf16* __restrict__ wt)
{
    const int t = blockIdx.x * blockDim.x + threadIdx.x;
    if (t >= 3 * DH * DMODEL) return;
    const int w   = t / (DH * DMODEL);
    const int rem = t - w * (DH * DMODEL);
    const int d   = rem / DMODEL;
    const int kk  = rem - d * DMODEL;
    const float* W = (w == 0) ? Wq : (w == 1) ? Wk : Wv;
    wt[t] = (__bf16)W[(size_t)kk * DH + d];
}

// =====================================================================
// Kernel 1: fused QKV projection, one wave per 16x16 output tile.
// A = bf16(x[16x32]) streamed from fp32; B = WT contiguous v16bf loads.
// Q,K stored row-major bf16; V stored TRANSPOSED (VT [B][64][S]) with a
// single packed 16-byte store per lane so attention reads contiguously.
// =====================================================================
__global__ __launch_bounds__(32) void qkv_proj_kernel(
    const float*  __restrict__ x,
    const __bf16* __restrict__ wt,      // [3][64][1024]
    __bf16* __restrict__ qo,            // [B*S,64]
    __bf16* __restrict__ ko,            // [B*S,64]
    __bf16* __restrict__ vto)           // [B][64][S]
{
    const int rowTile = blockIdx.x;                 // 0 .. B*S/16-1
    const int ct      = blockIdx.y;                 // 0 .. 11
    const int wsel    = ct >> 2;                    // 0=q 1=k 2=v
    const int col0    = (ct & 3) * 16;

    const int lane = threadIdx.x & 31;
    const int ln   = lane & 15;
    const int h    = lane >> 4;

    const float*  xrow = x  + (size_t)(rowTile * 16 + ln) * DMODEL;
    const __bf16* wcol = wt + (size_t)wsel * DH * DMODEL
                            + (size_t)(col0 + ln) * DMODEL + 16 * h;

    v8f acc = {};
    for (int k0 = 0; k0 < DMODEL; k0 += 32) {
        __builtin_prefetch(xrow + k0 + 64, 0, 1);   // global_prefetch_b8

        // A fragment: lane(m=ln,g=h): elem i -> K=16g+i
        v16bf a;
        const float* pa0 = xrow + k0 + 16 * h;
        const float* pa1 = xrow + k0 + 16 * h + 8;
        #pragma unroll
        for (int i = 0; i < 8; ++i) {
            a[i]     = (__bf16)pa0[i];
            a[8 + i] = (__bf16)pa1[i];
        }
        // B fragment: contiguous 32B from WT (elem i -> K=16h+i)
        v16bf bw = *(const v16bf*)(wcol + k0);

        acc = __builtin_amdgcn_wmma_f32_16x16x32_bf16(
                  false, a, false, bw, (short)0, acc, false, false);
    }

    if (wsel < 2) {
        __bf16* op   = (wsel == 0) ? qo : ko;
        __bf16* orow = op + (size_t)(rowTile * 16) * DH + col0 + ln;
        #pragma unroll
        for (int r = 0; r < 8; ++r)
            orow[(size_t)(r + 8 * h) * DH] = (__bf16)acc[r];
    } else {
        // transposed store: rows m=r+8h are contiguous along S
        const int bb = (rowTile * 16) / S_LEN;
        const int s0 = (rowTile * 16) % S_LEN;
        v8bf pk;
        #pragma unroll
        for (int r = 0; r < 8; ++r) pk[r] = (__bf16)acc[r];
        *(v8bf*)(vto + ((size_t)bb * DH + col0 + ln) * S_LEN + s0 + 8 * h) = pk;
    }
}

// =====================================================================
// Kernel 2: causal flash attention, one wave per 16-query tile.
// Per 32-key block: K tile (32x64) and VT tile (64x32) DMA'd into LDS by
// the Tensor Data Mover (s_wait_tensorcnt sync), 4 WMMA scores, fp32
// online softmax, P restaged C->A layout via LDS, 4 WMMA for P@V.
// =====================================================================
__global__ __launch_bounds__(32) void flash_attn_kernel(
    const __bf16* __restrict__ qg,      // [B*S,64]
    const __bf16* __restrict__ kg,      // [B*S,64]
    const __bf16* __restrict__ vtg,     // [B][64][S]
    const unsigned char* __restrict__ mask,
    float* __restrict__ out)
{
    __shared__ __attribute__((aligned(32))) __bf16 Klds[32 * DH];   // keys x d
    __shared__ __attribute__((aligned(32))) __bf16 VTlds[DH * 32];  // d x keys
    __shared__ __attribute__((aligned(32))) __bf16 Plds[16 * 32];

    const int tile = blockIdx.x;
    const int b    = tile >> 7;                  // / (S/16)
    const int q0   = (tile & 127) * 16;

    const int lane = threadIdx.x & 31;
    const int ln   = lane & 15;
    const int h    = lane >> 4;

    const __bf16* qbase  = qg  + (size_t)b * S_LEN * DH;
    const __bf16* kbase  = kg  + (size_t)b * S_LEN * DH;
    const __bf16* vtbase = vtg + (size_t)b * DH * S_LEN;
    const unsigned char* mb = mask + (size_t)b * S_LEN;

    // Q A-fragments (d=0:32, 32:64), packed 16B loads + shuffle
    v16bf aQ0, aQ1;
    {
        const __bf16* qr = qbase + (size_t)(q0 + ln) * DH;
        v8bf q00 = *(const v8bf*)(qr + 16 * h);
        v8bf q01 = *(const v8bf*)(qr + 16 * h + 8);
        v8bf q10 = *(const v8bf*)(qr + 32 + 16 * h);
        v8bf q11 = *(const v8bf*)(qr + 32 + 16 * h + 8);
        aQ0 = __builtin_shufflevector(q00, q01, 0,1,2,3,4,5,6,7,8,9,10,11,12,13,14,15);
        aQ1 = __builtin_shufflevector(q10, q11, 0,1,2,3,4,5,6,7,8,9,10,11,12,13,14,15);
    }

    v8f o[4] = {};
    float mrow[8], lsum[8];
    #pragma unroll
    for (int r = 0; r < 8; ++r) { mrow[r] = -3.0e38f; lsum[r] = 0.0f; }

    const int kbMax = (q0 + 15) / 32;            // inclusive causal bound
    for (int kbi = 0; kbi <= kbMax; ++kbi) {
        const int kv0 = kbi * 32;

        // ---- stage K (32x64) and VT (64x32) tiles into LDS ----
#if USE_TDM
        tdm_load_tile_2d((unsigned)(unsigned long long)(void*)Klds,
                         (const void*)(kbase + (size_t)kv0 * DH),
                         DH, S_LEN, /*tile*/ DH, 32, /*stride*/ DH);
        tdm_load_tile_2d((unsigned)(unsigned long long)(void*)VTlds,
                         (const void*)(vtbase + kv0),
                         S_LEN, DH, /*tile*/ 32, DH, /*stride*/ S_LEN);
        __builtin_amdgcn_s_wait_tensorcnt(0);
#else
        #pragma unroll
        for (int e = lane; e < (32 * DH) / 8; e += 32) {
            const int key = e >> 3, ch = e & 7;
            *(v8bf*)&Klds[key * DH + ch * 8] =
                *(const v8bf*)&kbase[(size_t)(kv0 + key) * DH + ch * 8];
        }
        #pragma unroll
        for (int e = lane; e < (DH * 32) / 8; e += 32) {
            const int d = e >> 2, ch = e & 3;
            *(v8bf*)&VTlds[d * 32 + ch * 8] =
                *(const v8bf*)&vtbase[(size_t)d * S_LEN + kv0 + ch * 8];
        }
#endif

        // ---- K B-fragments: contiguous 32B LDS reads ----
        v16bf bK00 = *(const v16bf*)&Klds[(ln)      * DH + 16 * h];
        v16bf bK01 = *(const v16bf*)&Klds[(ln)      * DH + 32 + 16 * h];
        v16bf bK10 = *(const v16bf*)&Klds[(16 + ln) * DH + 16 * h];
        v16bf bK11 = *(const v16bf*)&Klds[(16 + ln) * DH + 32 + 16 * h];

        v8f s0 = {}, s1 = {};
        s0 = __builtin_amdgcn_wmma_f32_16x16x32_bf16(false, aQ0, false, bK00, (short)0, s0, false, false);
        s0 = __builtin_amdgcn_wmma_f32_16x16x32_bf16(false, aQ1, false, bK01, (short)0, s0, false, false);
        s1 = __builtin_amdgcn_wmma_f32_16x16x32_bf16(false, aQ0, false, bK10, (short)0, s1, false, false);
        s1 = __builtin_amdgcn_wmma_f32_16x16x32_bf16(false, aQ1, false, bK11, (short)0, s1, false, false);

        // ---- scale + causal/key-padding mask (C layout) ----
        const int j0 = kv0 + ln, j1 = kv0 + 16 + ln;
        const bool km0 = mb[j0] != 0, km1 = mb[j1] != 0;
        float p0[8], p1[8], mloc[8];
        #pragma unroll
        for (int r = 0; r < 8; ++r) {
            const int qi = q0 + r + 8 * h;
            const float v0 = (km0 && (j0 <= qi)) ? s0[r] * 0.125f : -1.0e30f;
            const float v1 = (km1 && (j1 <= qi)) ? s1[r] * 0.125f : -1.0e30f;
            p0[r] = v0; p1[r] = v1;
            mloc[r] = fmaxf(v0, v1);
        }
        #pragma unroll
        for (int d = 1; d < 16; d <<= 1)
            #pragma unroll
            for (int r = 0; r < 8; ++r)
                mloc[r] = fmaxf(mloc[r], __shfl_xor(mloc[r], d, 32));

        float scl[8], rsum[8];
        #pragma unroll
        for (int r = 0; r < 8; ++r) {
            const float mnew = fmaxf(mrow[r], mloc[r]);
            scl[r]  = __expf(mrow[r] - mnew);
            mrow[r] = mnew;
            p0[r]   = __expf(p0[r] - mnew);
            p1[r]   = __expf(p1[r] - mnew);
            rsum[r] = p0[r] + p1[r];
        }
        #pragma unroll
        for (int d = 1; d < 16; d <<= 1)
            #pragma unroll
            for (int r = 0; r < 8; ++r)
                rsum[r] += __shfl_xor(rsum[r], d, 32);
        #pragma unroll
        for (int r = 0; r < 8; ++r) {
            lsum[r] = lsum[r] * scl[r] + rsum[r];
            o[0][r] *= scl[r]; o[1][r] *= scl[r];
            o[2][r] *= scl[r]; o[3][r] *= scl[r];
        }

        // ---- stage P (C layout -> LDS 16x32 row-major) ----
        #pragma unroll
        for (int r = 0; r < 8; ++r) {
            Plds[(r + 8 * h) * 32 + ln]      = (__bf16)p0[r];
            Plds[(r + 8 * h) * 32 + 16 + ln] = (__bf16)p1[r];
        }
        // same-wave DS ops are in-order; compiler waits dscnt before use.

        // ---- P A-fragment (16 x 32 keys) ----
        v8bf plo = *(const v8bf*)&Plds[ln * 32 + 16 * h];
        v8bf phi = *(const v8bf*)&Plds[ln * 32 + 16 * h + 8];
        v16bf aP = __builtin_shufflevector(plo, phi, 0,1,2,3,4,5,6,7,8,9,10,11,12,13,14,15);

        // ---- O += P @ V : 4 d-tiles, contiguous VT LDS fragments ----
        #pragma unroll
        for (int t = 0; t < 4; ++t) {
            v16bf bV = *(const v16bf*)&VTlds[(16 * t + ln) * 32 + 16 * h];
            o[t] = __builtin_amdgcn_wmma_f32_16x16x32_bf16(
                       false, aP, false, bV, (short)0, o[t], false, false);
        }
    }

    // ---- epilogue ----
    float inv[8];
    #pragma unroll
    for (int r = 0; r < 8; ++r) inv[r] = 1.0f / lsum[r];

    float* orow = out + ((size_t)b * S_LEN + q0) * DH;
    #pragma unroll
    for (int t = 0; t < 4; ++t)
        #pragma unroll
        for (int r = 0; r < 8; ++r)
            orow[(size_t)(r + 8 * h) * DH + 16 * t + ln] = o[t][r] * inv[r];
}

// =====================================================================
// Host launcher
// =====================================================================
extern "C" void kernel_launch(void* const* d_in, const int* in_sizes, int n_in,
                              void* d_out, int out_size, void* d_ws, size_t ws_size,
                              hipStream_t stream) {
    const float* x  = (const float*)d_in[0];
    const unsigned char* mask = (const unsigned char*)d_in[1]; // jnp bool
    const float* Wq = (const float*)d_in[2];
    const float* Wk = (const float*)d_in[3];
    const float* Wv = (const float*)d_in[4];
    float* out = (float*)d_out;

    // workspace: q, k (row-major), vT (transposed), wT   -- all bf16
    const size_t per = (size_t)NBATCH * S_LEN * DH;     // 524288 elems
    __bf16* qws = (__bf16*)d_ws;
    __bf16* kws = qws + per;
    __bf16* vtw = kws + per;
    __bf16* wtw = vtw + per;                            // 3*64*1024 elems

    const int wtN = 3 * DH * DMODEL;
    wtrans_kernel<<<dim3((wtN + 255) / 256), dim3(256), 0, stream>>>(
        Wq, Wk, Wv, wtw);

    const int rowTiles = (NBATCH * S_LEN) / 16;         // 512
    qkv_proj_kernel<<<dim3(rowTiles, 12), dim3(32), 0, stream>>>(
        x, wtw, qws, kws, vtw);

    flash_attn_kernel<<<dim3(rowTiles), dim3(32), 0, stream>>>(
        qws, kws, vtw, mask, out);
}
